// FireSpreadGNN_2637109920209
// MI455X (gfx1250) — matrix operand order repaired and checked by reading, not compiled, not run on hardware
//
#include <hip/hip_runtime.h>
#include <hip/hip_bf16.h>
#include <math.h>

#define NN 50000
#define EE 800000
#define NF 16
#define EF 8
#define HD 64
#define HEADS 4

typedef __attribute__((ext_vector_type(16))) _Float16 v16h;
typedef __attribute__((ext_vector_type(8)))  _Float16 v8h;
typedef __attribute__((ext_vector_type(8)))  float    v8f;

__device__ __forceinline__ float leaky02(float x) { return x > 0.f ? x : 0.2f * x; }
__device__ __forceinline__ unsigned fenc(float f) {
    unsigned u = __float_as_uint(f);
    return (u & 0x80000000u) ? ~u : (u | 0x80000000u);
}
__device__ __forceinline__ float fdec(unsigned u) {
    return (u & 0x80000000u) ? __uint_as_float(u & 0x7FFFFFFFu) : __uint_as_float(~u);
}

// ---------------------------------------------------------------- fill
__global__ void k_fill_u32(unsigned* p, unsigned v, size_t n) {
    size_t i = (size_t)blockIdx.x * 256u + threadIdx.x;
    if (i < n) p[i] = v;
}

// ---------------------------------------------------------------- derived edge-attention vectors
// u1[8][4], c1[4]  :  a_e1[e,h] = edge_attr[e]·u1[:,h] + c1[h]
// u2[8],    c2     :  a_e2[e]   = edge_attr[e]·u2 + c2
__global__ void k_derive(const float* __restrict__ g1_eW, const float* __restrict__ g1_ae,
                         const float* __restrict__ g2_eW, const float* __restrict__ g2_ae,
                         const float* __restrict__ edge_W, const float* __restrict__ edge_b,
                         float* __restrict__ u1, float* __restrict__ c1,
                         float* __restrict__ u2, float* __restrict__ c2) {
    __shared__ float w1e[64][4];
    __shared__ float w2e[64];
    int t = threadIdx.x;                    // 64 threads
    for (int h = 0; h < 4; ++h) {
        float s = 0.f;
        for (int c = 0; c < 64; ++c) s += g1_eW[t * 256 + h * 64 + c] * g1_ae[h * 64 + c];
        w1e[t][h] = s;
    }
    { float s = 0.f; for (int c = 0; c < 64; ++c) s += g2_eW[t * 64 + c] * g2_ae[c]; w2e[t] = s; }
    __syncthreads();
    if (t < 8) {
        for (int h = 0; h < 4; ++h) {
            float s = 0.f;
            for (int k = 0; k < 64; ++k) s += edge_W[t * 64 + k] * w1e[k][h];
            u1[t * 4 + h] = s;
        }
        float s2 = 0.f;
        for (int k = 0; k < 64; ++k) s2 += edge_W[t * 64 + k] * w2e[k];
        u2[t] = s2;
    }
    if (t < 4) { float s = 0.f; for (int k = 0; k < 64; ++k) s += edge_b[k] * w1e[k][t]; c1[t] = s; }
    if (t == 0) { float s = 0.f; for (int k = 0; k < 64; ++k) s += edge_b[k] * w2e[k]; *c2 = s; }
}

// ---------------------------------------------------------------- generic WMMA GEMM (specialized)
// OUT[M, Ntot] (cols [c0,c0+64)) = act(A @ B + bias); 128 threads = 4 waves, each wave one
// 16x16 tile. GATHER: A-row e = concat(h[src[e]], h[dst[e]]) (KREAL=128). REDUCE: fold
// sum_c relu(acc+bias[c])*w2[c] into redout[row] via cross-lane reduction + atomicAdd.
template <int KREAL, bool GATHER, bool REDUCE, bool ACT, bool BIAS>
__global__ __launch_bounds__(128)
void k_wmma_gemm(const float* __restrict__ A, const int* __restrict__ gsrc,
                 const int* __restrict__ gdst, const float* __restrict__ B,
                 const float* __restrict__ bias, float* __restrict__ OUT,
                 float* __restrict__ redout, const float* __restrict__ w2, int Ntot) {
    constexpr int KPAD = (KREAL + 31) & ~31;
    constexpr int LDK  = KREAL + 8;           // halves per col (pad: conflict-free, 16B aligned)
    __shared__ _Float16 sBt[64 * LDK];        // B staged transposed: sBt[c][k]

    const int c0  = blockIdx.y * 64;
    const int tid = threadIdx.x;
    for (int i = tid; i < KREAL * 64; i += 128) {
        int k = i >> 6, c = i & 63;
        sBt[c * LDK + k] = (_Float16)B[k * Ntot + c0 + c];
    }
    __syncthreads();

    const int wave  = tid >> 5, lane = tid & 31;
    const int row0  = blockIdx.x * 16;
    const int lrow  = row0 + (lane & 15);     // A-row this lane supplies
    const int kbase = (lane >> 4) * 8;

    const float* rowA0;
    const float* rowA1;
    if (GATHER) {
        rowA0 = A + (size_t)gsrc[lrow] * HD;
        rowA1 = A + (size_t)gdst[lrow] * HD;
    } else {
        rowA0 = A + (size_t)lrow * KREAL;
        rowA1 = rowA0;
    }

    const _Float16* bcol = sBt + (wave * 16 + (lane & 15)) * LDK;
    const int koff = (lane >> 4) * 16;

    v8f acc = {0.f, 0.f, 0.f, 0.f, 0.f, 0.f, 0.f, 0.f};
#pragma unroll
    for (int kc = 0; kc < KPAD; kc += 32) {
        v16h a;
        // chunk0: K = kc+kbase+0..7 (never straddles the 64-split; side fixed by kc)
        if (kc + 16 <= KREAL) {
            const float* p = (GATHER && (kc >= HD)) ? (rowA1 - HD) : rowA0;
            const int kk = kc + kbase;
            float4 q0 = *(const float4*)(p + kk);
            float4 q1 = *(const float4*)(p + kk + 4);
            a[0] = (_Float16)q0.x; a[1] = (_Float16)q0.y;
            a[2] = (_Float16)q0.z; a[3] = (_Float16)q0.w;
            a[4] = (_Float16)q1.x; a[5] = (_Float16)q1.y;
            a[6] = (_Float16)q1.z; a[7] = (_Float16)q1.w;
        } else {
#pragma unroll
            for (int h = 0; h < 8; ++h) a[h] = (_Float16)0.f;
        }
        // chunk1: K = kc+kbase+16..23
        if (kc + 32 <= KREAL) {
            const float* p = (GATHER && (kc + 16 >= HD)) ? (rowA1 - HD) : rowA0;
            const int kk = kc + kbase + 16;
            float4 q2 = *(const float4*)(p + kk);
            float4 q3 = *(const float4*)(p + kk + 4);
            a[8]  = (_Float16)q2.x; a[9]  = (_Float16)q2.y;
            a[10] = (_Float16)q2.z; a[11] = (_Float16)q2.w;
            a[12] = (_Float16)q3.x; a[13] = (_Float16)q3.y;
            a[14] = (_Float16)q3.z; a[15] = (_Float16)q3.w;
        } else {
#pragma unroll
            for (int h = 0; h < 8; ++h) a[h + 8] = (_Float16)0.f;
        }
        // B fragment: K = kc+kbase+0..7 and kc+kbase+16..23 for this col (two 16B LDS loads)
        v16h b;
        {
            const _Float16* bp = bcol + kc + kbase;      // 16B aligned (LDK even)
            v8h b0 = *(const v8h*)(bp);
            v8h b1 = {};
            if (kc + 32 <= KREAL) b1 = *(const v8h*)(bp + 16);
#pragma unroll
            for (int h = 0; h < 8; ++h) { b[h] = b0[h]; b[h + 8] = b1[h]; }
        }
        acc = __builtin_amdgcn_wmma_f32_16x16x32_f16(false, a, false, b, (short)0, acc,
                                                     false, false);
    }

    const int colg  = c0 + wave * 16 + (lane & 15);
    const int rbase = row0 + (lane >> 4) * 8;
    const float bv  = BIAS ? bias[colg] : 0.f;           // hoisted: one load, no per-r guard
    if (!REDUCE) {
        float* op = OUT + (size_t)rbase * Ntot + colg;
#pragma unroll
        for (int r = 0; r < 8; ++r) {
            float v = acc[r] + bv;
            if (ACT) v = fmaxf(v, 0.f);
            op[(size_t)r * Ntot] = v;
        }
    } else {
        float s[8];
        const float wv = w2[colg];
#pragma unroll
        for (int r = 0; r < 8; ++r) s[r] = fmaxf(acc[r] + bv, 0.f) * wv;
#pragma unroll
        for (int off = 1; off < 16; off <<= 1)
#pragma unroll
            for (int r = 0; r < 8; ++r) s[r] += __shfl_xor(s[r], off, 32);
        if ((lane & 15) == 0)
#pragma unroll
            for (int r = 0; r < 8; ++r) atomicAdd(&redout[rbase + r], s[r]);
    }
}

// ---------------------------------------------------------------- degree + raw edge-attr segment sum
__global__ void k_deg_sumea(const int* __restrict__ dst, const float* __restrict__ ea,
                            float* __restrict__ deg, float* __restrict__ sumea) {
    int e = blockIdx.x * 256 + threadIdx.x;
    if (e >= EE) return;
    int d = dst[e];
    atomicAdd(&deg[d], 1.f);
#pragma unroll
    for (int j = 0; j < EF; ++j) atomicAdd(&sumea[d * EF + j], ea[e * EF + j]);
}

// ---------------------------------------------------------------- layer-1 attention scalars
__global__ void k_attn_nd1(const float* __restrict__ ht1, const float* __restrict__ asw,
                           const float* __restrict__ adw, float* __restrict__ as1,
                           float* __restrict__ ad1) {
    int i = blockIdx.x * 256 + threadIdx.x;
    if (i >= NN * HEADS) return;
    int n = i >> 2, h = i & 3;
    const float* r = ht1 + (size_t)n * 256 + h * 64;
    float s = 0.f, d = 0.f;
    for (int c = 0; c < 64; ++c) { float t = r[c]; s += t * asw[h * 64 + c]; d += t * adw[h * 64 + c]; }
    as1[i] = s; ad1[i] = d;
}

__global__ void k_logit1(const int* __restrict__ src, const int* __restrict__ dst,
                         const float* __restrict__ ea, const float* __restrict__ u1,
                         const float* __restrict__ c1, const float* __restrict__ as1,
                         const float* __restrict__ ad1, float* __restrict__ w1,
                         unsigned* __restrict__ mx1) {
    int e = blockIdx.x * 256 + threadIdx.x;
    if (e >= EE) return;
    int s = src[e], d = dst[e];
    float v[EF];
#pragma unroll
    for (int j = 0; j < EF; ++j) v[j] = ea[e * EF + j];
#pragma unroll
    for (int h = 0; h < 4; ++h) {
        float aeh = c1[h];
#pragma unroll
        for (int j = 0; j < EF; ++j) aeh += v[j] * u1[j * 4 + h];
        float l = leaky02(as1[s * 4 + h] + ad1[d * 4 + h] + aeh);
        w1[e * 4 + h] = l;
        atomicMax(&mx1[d * 4 + h], fenc(l));
    }
}

__global__ void k_logit1_loop(const float* __restrict__ deg, const float* __restrict__ sumea,
                              const float* __restrict__ u1, const float* __restrict__ c1,
                              const float* __restrict__ as1, const float* __restrict__ ad1,
                              float* __restrict__ wl1, unsigned* __restrict__ mx1) {
    int n = blockIdx.x * 256 + threadIdx.x;
    if (n >= NN) return;
    float dg = deg[n], inv = 1.f / fmaxf(dg, 1.f);
    float se[EF];
#pragma unroll
    for (int j = 0; j < EF; ++j) se[j] = sumea[n * EF + j];
#pragma unroll
    for (int h = 0; h < 4; ++h) {
        float aeh = dg * c1[h];
#pragma unroll
        for (int j = 0; j < EF; ++j) aeh += se[j] * u1[j * 4 + h];
        aeh *= inv;
        float l = leaky02(as1[n * 4 + h] + ad1[n * 4 + h] + aeh);
        wl1[n * 4 + h] = l;
        atomicMax(&mx1[n * 4 + h], fenc(l));
    }
}

__global__ void k_exp1(const int* __restrict__ dst, float* __restrict__ w1,
                       const unsigned* __restrict__ mx1, float* __restrict__ den1) {
    int i = blockIdx.x * 256 + threadIdx.x;
    if (i >= EE * HEADS) return;
    int e = i >> 2, h = i & 3;
    int d = dst[e];
    float w = __expf(w1[i] - fdec(mx1[d * 4 + h]));
    w1[i] = w;
    atomicAdd(&den1[d * 4 + h], w);
}

__global__ void k_exp1_loop(float* __restrict__ wl1, const unsigned* __restrict__ mx1,
                            float* __restrict__ den1) {
    int i = blockIdx.x * 256 + threadIdx.x;
    if (i >= NN * HEADS) return;
    float w = __expf(wl1[i] - fdec(mx1[i]));
    wl1[i] = w;
    atomicAdd(&den1[i], w);
}

__global__ void k_agg1(const int* __restrict__ src, const int* __restrict__ dst,
                       const float* __restrict__ w1, const float* __restrict__ den1,
                       const float* __restrict__ ht1, float* __restrict__ acc1) {
    int i = blockIdx.x * 256 + threadIdx.x;
    if (i >= EE * HEADS) return;
    int e = i >> 2, h = i & 3;
    int s = src[e], d = dst[e];
    float a = w1[i] / (den1[d * 4 + h] + 1e-16f);
    const float* hr = ht1 + (size_t)s * 256 + h * 64;
    float* o = acc1 + (size_t)d * 256 + h * 64;
#pragma unroll 8
    for (int c = 0; c < 64; ++c) atomicAdd(&o[c], a * hr[c]);
}

__global__ void k_agg1_loop(const float* __restrict__ wl1, const float* __restrict__ den1,
                            const float* __restrict__ ht1, float* __restrict__ acc1) {
    int i = blockIdx.x * 256 + threadIdx.x;
    if (i >= NN * HEADS) return;
    int n = i >> 2, h = i & 3;
    float a = wl1[i] / (den1[i] + 1e-16f);
    const float* hr = ht1 + (size_t)n * 256 + h * 64;
    float* o = acc1 + (size_t)n * 256 + h * 64;
#pragma unroll 8
    for (int c = 0; c < 64; ++c) atomicAdd(&o[c], a * hr[c]);
}

__global__ void k_bias_relu(float* __restrict__ x, const float* __restrict__ b,
                            unsigned mask, size_t n) {
    size_t i = (size_t)blockIdx.x * 256u + threadIdx.x;
    if (i >= n) return;
    x[i] = fmaxf(x[i] + b[i & mask], 0.f);
}

// ---------------------------------------------------------------- layer-2 attention (1 head)
__global__ void k_attn_nd2(const float* __restrict__ ht2, const float* __restrict__ asw,
                           const float* __restrict__ adw, float* __restrict__ as2,
                           float* __restrict__ ad2) {
    int n = blockIdx.x * 256 + threadIdx.x;
    if (n >= NN) return;
    const float* r = ht2 + (size_t)n * 64;
    float s = 0.f, d = 0.f;
    for (int c = 0; c < 64; ++c) { float t = r[c]; s += t * asw[c]; d += t * adw[c]; }
    as2[n] = s; ad2[n] = d;
}

__global__ void k_logit2(const int* __restrict__ src, const int* __restrict__ dst,
                         const float* __restrict__ ea, const float* __restrict__ u2,
                         const float* __restrict__ c2, const float* __restrict__ as2,
                         const float* __restrict__ ad2, float* __restrict__ wb,
                         unsigned* __restrict__ mx2) {
    int e = blockIdx.x * 256 + threadIdx.x;
    if (e >= EE) return;
    int s = src[e], d = dst[e];
    float aeh = *c2;
#pragma unroll
    for (int j = 0; j < EF; ++j) aeh += ea[e * EF + j] * u2[j];
    float l = leaky02(as2[s] + ad2[d] + aeh);
    wb[e] = l;
    atomicMax(&mx2[d], fenc(l));
}

__global__ void k_logit2_loop(const float* __restrict__ deg, const float* __restrict__ sumea,
                              const float* __restrict__ u2, const float* __restrict__ c2,
                              const float* __restrict__ as2, const float* __restrict__ ad2,
                              float* __restrict__ wl2, unsigned* __restrict__ mx2) {
    int n = blockIdx.x * 256 + threadIdx.x;
    if (n >= NN) return;
    float dg = deg[n], inv = 1.f / fmaxf(dg, 1.f);
    float aeh = dg * (*c2);
#pragma unroll
    for (int j = 0; j < EF; ++j) aeh += sumea[n * EF + j] * u2[j];
    aeh *= inv;
    float l = leaky02(as2[n] + ad2[n] + aeh);
    wl2[n] = l;
    atomicMax(&mx2[n], fenc(l));
}

__global__ void k_exp2(const int* __restrict__ dst, float* __restrict__ wb,
                       const unsigned* __restrict__ mx2, float* __restrict__ den2) {
    int e = blockIdx.x * 256 + threadIdx.x;
    if (e >= EE) return;
    int d = dst[e];
    float w = __expf(wb[e] - fdec(mx2[d]));
    wb[e] = w;
    atomicAdd(&den2[d], w);
}

__global__ void k_exp2_loop(float* __restrict__ wl2, const unsigned* __restrict__ mx2,
                            float* __restrict__ den2) {
    int n = blockIdx.x * 256 + threadIdx.x;
    if (n >= NN) return;
    float w = __expf(wl2[n] - fdec(mx2[n]));
    wl2[n] = w;
    atomicAdd(&den2[n], w);
}

__global__ void k_agg2(const int* __restrict__ src, const int* __restrict__ dst,
                       const float* __restrict__ wb, const float* __restrict__ den2,
                       const float* __restrict__ ht2, float* __restrict__ acc2) {
    int e = blockIdx.x * 256 + threadIdx.x;
    if (e >= EE) return;
    int s = src[e], d = dst[e];
    float a = wb[e] / (den2[d] + 1e-16f);
    const float* hr = ht2 + (size_t)s * 64;
    float* o = acc2 + (size_t)d * 64;
#pragma unroll 8
    for (int c = 0; c < 64; ++c) atomicAdd(&o[c], a * hr[c]);
}

__global__ void k_agg2_loop(const float* __restrict__ wl2, const float* __restrict__ den2,
                            const float* __restrict__ ht2, float* __restrict__ acc2) {
    int n = blockIdx.x * 256 + threadIdx.x;
    if (n >= NN) return;
    float a = wl2[n] / (den2[n] + 1e-16f);
    const float* hr = ht2 + (size_t)n * 64;
    float* o = acc2 + (size_t)n * 64;
#pragma unroll 8
    for (int c = 0; c < 64; ++c) atomicAdd(&o[c], a * hr[c]);
}

__global__ void k_sig(const float* __restrict__ a, const float* __restrict__ b,
                      float* __restrict__ out, int n) {
    int i = blockIdx.x * 256 + threadIdx.x;
    if (i >= n) return;
    out[i] = 1.f / (1.f + __expf(-(a[i] + b[0])));
}

// ================================================================ host
extern "C" void kernel_launch(void* const* d_in, const int* in_sizes, int n_in,
                              void* d_out, int out_size, void* d_ws, size_t ws_size,
                              hipStream_t stream) {
    // setup_inputs order: x, edge_index, edge_attr, then params flattened in
    // sorted-key (jax pytree dict) order.
    const float* x      = (const float*)d_in[0];
    const int*   src    = (const int*)d_in[1];
    const int*   dst    = src + EE;
    const float* eattr  = (const float*)d_in[2];
    const float* edge_W = (const float*)d_in[3];
    const float* edge_b = (const float*)d_in[4];
    const float* ep_W1  = (const float*)d_in[5];
    const float* ep_W2  = (const float*)d_in[6];
    const float* ep_b1  = (const float*)d_in[7];
    const float* ep_b2  = (const float*)d_in[8];
    const float* g1_W   = (const float*)d_in[9];
    const float* g1_ad  = (const float*)d_in[10];
    const float* g1_ae  = (const float*)d_in[11];
    const float* g1_as  = (const float*)d_in[12];
    const float* g1_b   = (const float*)d_in[13];
    const float* g1_eW  = (const float*)d_in[14];
    const float* g2_W   = (const float*)d_in[15];
    const float* g2_ad  = (const float*)d_in[16];
    const float* g2_ae  = (const float*)d_in[17];
    const float* g2_as  = (const float*)d_in[18];
    const float* g2_b   = (const float*)d_in[19];
    const float* g2_eW  = (const float*)d_in[20];
    const float* node_W = (const float*)d_in[21];
    const float* node_b = (const float*)d_in[22];
    const float* np_W1  = (const float*)d_in[23];
    const float* np_W2  = (const float*)d_in[24];
    const float* np_b1  = (const float*)d_in[25];
    const float* np_b2  = (const float*)d_in[26];

    float* ws = (float*)d_ws;
    size_t o = 0;
    float*    h0    = ws + o; o += (size_t)NN * 64;
    float*    ht1   = ws + o; o += (size_t)NN * 256;
    float*    as1   = ws + o; o += (size_t)NN * 4;
    float*    ad1   = ws + o; o += (size_t)NN * 4;
    float*    deg   = ws + o; o += NN;
    float*    sumea = ws + o; o += (size_t)NN * 8;
    unsigned* mx1   = (unsigned*)(ws + o); o += (size_t)NN * 4;
    float*    den1  = ws + o; o += (size_t)NN * 4;
    float*    w1    = ws + o; o += (size_t)EE * 4;
    float*    wl1   = ws + o; o += (size_t)NN * 4;
    float*    acc1  = ws + o; o += (size_t)NN * 256;   // becomes h1 in place
    float*    ht2   = ws + o; o += (size_t)NN * 64;
    float*    as2   = ws + o; o += NN;
    float*    ad2   = ws + o; o += NN;
    unsigned* mx2   = (unsigned*)(ws + o); o += NN;
    float*    den2  = ws + o; o += NN;
    float*    wb2   = ws + o; o += EE;
    float*    wl2   = ws + o; o += NN;
    float*    acc2  = ws + o; o += (size_t)NN * 64;    // becomes h2 in place
    float*    epred = ws + o; o += EE;
    float*    npred = ws + o; o += NN;
    float*    u1    = ws + o; o += 32;
    float*    c1    = ws + o; o += 4;
    float*    u2    = ws + o; o += 8;
    float*    c2    = ws + o; o += 4;

    float* nout = (float*)d_out;       // node_pred [N] then edge_pred [E]
    float* eout = nout + NN;

    const unsigned NEG_ENC = 0x007FFFFFu;  // fenc(-inf)
    auto grd = [](size_t n) { return dim3((unsigned)((n + 255) / 256)); };

    // ---- zero / init scratch (deterministic every call)
    k_fill_u32<<<grd(NN), 256, 0, stream>>>((unsigned*)deg, 0u, NN);
    k_fill_u32<<<grd((size_t)NN * 8), 256, 0, stream>>>((unsigned*)sumea, 0u, (size_t)NN * 8);
    k_fill_u32<<<grd((size_t)NN * 4), 256, 0, stream>>>(mx1, NEG_ENC, (size_t)NN * 4);
    k_fill_u32<<<grd((size_t)NN * 4), 256, 0, stream>>>((unsigned*)den1, 0u, (size_t)NN * 4);
    k_fill_u32<<<grd((size_t)NN * 256), 256, 0, stream>>>((unsigned*)acc1, 0u, (size_t)NN * 256);
    k_fill_u32<<<grd(NN), 256, 0, stream>>>(mx2, NEG_ENC, NN);
    k_fill_u32<<<grd(NN), 256, 0, stream>>>((unsigned*)den2, 0u, NN);
    k_fill_u32<<<grd((size_t)NN * 64), 256, 0, stream>>>((unsigned*)acc2, 0u, (size_t)NN * 64);
    k_fill_u32<<<grd(EE), 256, 0, stream>>>((unsigned*)epred, 0u, EE);
    k_fill_u32<<<grd(NN), 256, 0, stream>>>((unsigned*)npred, 0u, NN);

    // ---- derived edge-attention projections + degree / raw edge-attr segment sums
    k_derive<<<1, 64, 0, stream>>>(g1_eW, g1_ae, g2_eW, g2_ae, edge_W, edge_b, u1, c1, u2, c2);
    k_deg_sumea<<<grd(EE), 256, 0, stream>>>(dst, eattr, deg, sumea);

    // ---- h0 = x @ node_W + node_b          [N,16]x[16,64]
    k_wmma_gemm<NF, false, false, false, true><<<dim3(NN / 16, 1), 128, 0, stream>>>(
        x, nullptr, nullptr, node_W, node_b, h0, nullptr, nullptr, 64);
    // ---- ht1 = h0 @ g1_W                   [N,64]x[64,256]
    k_wmma_gemm<64, false, false, false, false><<<dim3(NN / 16, 4), 128, 0, stream>>>(
        h0, nullptr, nullptr, g1_W, nullptr, ht1, nullptr, nullptr, 256);
    // ---- layer-1 attention + scatter softmax + aggregation
    k_attn_nd1<<<grd((size_t)NN * 4), 256, 0, stream>>>(ht1, g1_as, g1_ad, as1, ad1);
    k_logit1<<<grd(EE), 256, 0, stream>>>(src, dst, eattr, u1, c1, as1, ad1, w1, mx1);
    k_logit1_loop<<<grd(NN), 256, 0, stream>>>(deg, sumea, u1, c1, as1, ad1, wl1, mx1);
    k_exp1<<<grd((size_t)EE * 4), 256, 0, stream>>>(dst, w1, mx1, den1);
    k_exp1_loop<<<grd((size_t)NN * 4), 256, 0, stream>>>(wl1, mx1, den1);
    k_agg1<<<grd((size_t)EE * 4), 256, 0, stream>>>(src, dst, w1, den1, ht1, acc1);
    k_agg1_loop<<<grd((size_t)NN * 4), 256, 0, stream>>>(wl1, den1, ht1, acc1);
    k_bias_relu<<<grd((size_t)NN * 256), 256, 0, stream>>>(acc1, g1_b, 255u, (size_t)NN * 256);

    // ---- ht2 = h1 @ g2_W                   [N,256]x[256,64]
    k_wmma_gemm<256, false, false, false, false><<<dim3(NN / 16, 1), 128, 0, stream>>>(
        acc1, nullptr, nullptr, g2_W, nullptr, ht2, nullptr, nullptr, 64);
    // ---- layer-2 attention (1 head)
    k_attn_nd2<<<grd(NN), 256, 0, stream>>>(ht2, g2_as, g2_ad, as2, ad2);
    k_logit2<<<grd(EE), 256, 0, stream>>>(src, dst, eattr, u2, c2, as2, ad2, wb2, mx2);
    k_logit2_loop<<<grd(NN), 256, 0, stream>>>(deg, sumea, u2, c2, as2, ad2, wl2, mx2);
    k_exp2<<<grd(EE), 256, 0, stream>>>(dst, wb2, mx2, den2);
    k_exp2_loop<<<grd(NN), 256, 0, stream>>>(wl2, mx2, den2);
    k_agg2<<<grd(EE), 256, 0, stream>>>(src, dst, wb2, den2, ht2, acc2);
    k_agg2_loop<<<grd(NN), 256, 0, stream>>>(wl2, den2, ht2, acc2);
    k_bias_relu<<<grd((size_t)NN * 64), 256, 0, stream>>>(acc2, g2_b, 63u, (size_t)NN * 64);

    // ---- edge predictor: relu(concat(h2[s],h2[d]) @ ep_W1 + b1) @ ep_W2, fused reduce
    k_wmma_gemm<128, true, true, true, true><<<dim3(EE / 16, 1), 128, 0, stream>>>(
        acc2, src, dst, ep_W1, ep_b1, nullptr, epred, ep_W2, 64);
    // ---- node predictor: relu(h2 @ np_W1 + b1) @ np_W2, fused reduce
    k_wmma_gemm<64, false, true, true, true><<<dim3(NN / 16, 1), 128, 0, stream>>>(
        acc2, nullptr, nullptr, np_W1, np_b1, nullptr, npred, np_W2, 64);

    // ---- sigmoids into d_out (node_pred first, then edge_pred)
    k_sig<<<grd(NN), 256, 0, stream>>>(npred, np_b2, nout, NN);
    k_sig<<<grd(EE), 256, 0, stream>>>(epred, ep_b2, eout, EE);
}
